// GCN_46428596470141
// MI455X (gfx1250) — matrix operand order repaired and checked by reading, not compiled, not run on hardware
//
#include <hip/hip_runtime.h>
#include <hip/hip_bf16.h>
#include <math.h>

typedef __attribute__((ext_vector_type(16))) _Float16 v16h;
typedef __attribute__((ext_vector_type(8)))  float    v8f;

#define FDIM   128
#define SCONST 1.0f   // s = 2 / lambda_max = 2/2 = 1  =>  tx1 = agg exactly

// ---------------------------------------------------------------- zero fill
__global__ void cheb_zero_kernel(float* __restrict__ p, size_t n) {
    size_t i      = (size_t)blockIdx.x * blockDim.x + threadIdx.x;
    size_t stride = (size_t)gridDim.x * blockDim.x;
    for (; i < n; i += stride) p[i] = 0.0f;
}

// ---------------------------------------------------------------- degree
__global__ void cheb_deg_kernel(const int* __restrict__ src,
                                float* __restrict__ deg, int E) {
    int i = blockIdx.x * blockDim.x + threadIdx.x;
    if (i < E) atomicAdd(&deg[src[i]], 1.0f);
}

__global__ void cheb_dinv_kernel(float* __restrict__ deg, int N) {
    int i = blockIdx.x * blockDim.x + threadIdx.x;
    if (i < N) {
        float d = deg[i];
        deg[i] = (d > 0.0f) ? rsqrtf(fmaxf(d, 1e-12f)) : 0.0f;
    }
}

// ---------------------------------------------------------------- edge norms
// norm[e] = -s * dinv[src] * dinv[dst]; computed once, used by both layers.
__global__ void cheb_norm_kernel(const int* __restrict__ src,
                                 const int* __restrict__ dst,
                                 const float* __restrict__ dinv,
                                 float* __restrict__ norm, int E) {
    int i = blockIdx.x * blockDim.x + threadIdx.x;
    if (i < E) norm[i] = -SCONST * dinv[src[i]] * dinv[dst[i]];
}

// ---------------------------------------------------------------- edge scatter
// One wave32 per edge; lane l handles features [4l, 4l+4).
// Wave reads feat row (512B fully coalesced), 4 f32 atomics per lane to dst row.
__global__ void cheb_agg_kernel(const float* __restrict__ feat,
                                const int* __restrict__ src,
                                const int* __restrict__ dst,
                                const float* __restrict__ norm,
                                float* __restrict__ out, int E) {
    int gid  = blockIdx.x * blockDim.x + threadIdx.x;
    int e    = gid >> 5;
    int lane = gid & 31;
    if (e >= E) return;
    int s = src[e], d = dst[e];
    float w = norm[e];
    const float4 v = *(const float4*)(feat + (size_t)s * FDIM + lane * 4);
    float* o = out + (size_t)d * FDIM + lane * 4;
    atomicAdd(o + 0, w * v.x);
    atomicAdd(o + 1, w * v.y);
    atomicAdd(o + 2, w * v.z);
    atomicAdd(o + 3, w * v.w);
}

// ---------------------------------------------------------------- layer-1 GEMM
// h = relu(X @ W0 + AGG @ W1 + bias)  (tx1 == agg since s == 1), WMMA f16.
// 8 waves per block, each wave: 16 rows x 128 cols (8 accumulator tiles).
__global__ __launch_bounds__(256) void cheb_gemm_relu_wmma(
        const float* __restrict__ X, const float* __restrict__ AGG,
        const float* __restrict__ W0, const float* __restrict__ W1,
        const float* __restrict__ Bias, float* __restrict__ Hout, int nTiles) {
    // B fragments pre-swizzled: [colTile][kstep][lane][16 halves] contiguous.
    __shared__ __align__(32) _Float16 lB0[8 * 4 * 32 * 16];
    __shared__ __align__(32) _Float16 lB1[8 * 4 * 32 * 16];

    for (int idx = threadIdx.x; idx < FDIM * FDIM; idx += blockDim.x) {
        int k = idx >> 7, n = idx & 127;          // W is [K=128][N=128] row-major
        int colTile = n >> 4, kstep = k >> 5;
        int kk = k & 31, nn = n & 15;
        int hsel  = (kk >> 3) & 1;                 // lane half: k-runs 8..15, 24..31 -> upper
        int lane2 = hsel * 16 + nn;
        int vreg  = ((kk & 7) >> 1) + ((kk >> 4) << 2);
        int par   = kk & 1;
        int pos   = ((colTile * 4 + kstep) * 32 + lane2) * 16 + vreg * 2 + par;
        lB0[pos] = (_Float16)W0[idx];
        lB1[pos] = (_Float16)W1[idx];
    }
    __syncthreads();

    int wave    = threadIdx.x >> 5;
    int lane    = threadIdx.x & 31;
    int rowTile = blockIdx.x * 8 + wave;
    if (rowTile >= nTiles) return;

    int m    = lane & 15;      // A-matrix row within tile (ISA 7.12.2 layout)
    int hsel = lane >> 4;      // K-half select
    size_t rowBase = ((size_t)rowTile * 16 + m) * FDIM;

    v8f acc[8];
#pragma unroll
    for (int c = 0; c < 8; ++c) acc[c] = (v8f)(0.0f);

#pragma unroll
    for (int t = 0; t < 4; ++t) {                  // K = 128 in steps of 32
        int kbase = t * 32 + hsel * 8;
        const float* px = X   + rowBase + kbase;
        const float* pa = AGG + rowBase + kbase;
        v16h aX, aT;
#pragma unroll
        for (int j = 0; j < 8; ++j) {              // runs K+0..7 and K+16..23
            aX[j]     = (_Float16)px[j];
            aT[j]     = (_Float16)pa[j];
            aX[8 + j] = (_Float16)px[16 + j];
            aT[8 + j] = (_Float16)pa[16 + j];
        }
#pragma unroll
        for (int c = 0; c < 8; ++c) {
            const v16h b0 = *(const v16h*)&lB0[((c * 4 + t) * 32 + lane) * 16];
            const v16h b1 = *(const v16h*)&lB1[((c * 4 + t) * 32 + lane) * 16];
            acc[c] = __builtin_amdgcn_wmma_f32_16x16x32_f16(
                         false, aX, false, b0, (short)0, acc[c], false, false);
            acc[c] = __builtin_amdgcn_wmma_f32_16x16x32_f16(
                         false, aT, false, b1, (short)0, acc[c], false, false);
        }
    }

    // C/D layout: lane n = lane&15, VGPR r -> row r + 8*(lane>>4)
#pragma unroll
    for (int c = 0; c < 8; ++c) {
        int col  = c * 16 + (lane & 15);
        float bb = Bias[col];
#pragma unroll
        for (int r = 0; r < 8; ++r) {
            int mm   = r + hsel * 8;
            float vv = acc[c][r] + bb;
            Hout[((size_t)rowTile * 16 + mm) * FDIM + col] = fmaxf(vv, 0.0f);
        }
    }
}

// ---------------------------------------------------------------- layer-2 head
// out[n] = sigmoid(h[n]·w0 + agg2[n]·w1 + b). One wave per node.
__global__ void cheb_out_kernel(const float* __restrict__ Hf,
                                const float* __restrict__ AGG2,
                                const float* __restrict__ w0,
                                const float* __restrict__ w1,
                                const float* __restrict__ b2,
                                float* __restrict__ out, int N) {
    int gid  = blockIdx.x * blockDim.x + threadIdx.x;
    int node = gid >> 5;
    int lane = gid & 31;
    if (node >= N) return;
    const float4 hv  = *(const float4*)(Hf   + (size_t)node * FDIM + lane * 4);
    const float4 av  = *(const float4*)(AGG2 + (size_t)node * FDIM + lane * 4);
    const float4 w0v = *(const float4*)(w0 + lane * 4);
    const float4 w1v = *(const float4*)(w1 + lane * 4);
    float p = hv.x * w0v.x + hv.y * w0v.y + hv.z * w0v.z + hv.w * w0v.w
            + av.x * w1v.x + av.y * w1v.y + av.z * w1v.z + av.w * w1v.w;
#pragma unroll
    for (int off = 16; off > 0; off >>= 1) p += __shfl_xor(p, off, 32);
    if (lane == 0) out[node] = 1.0f / (1.0f + expf(-(p + b2[0])));
}

// ---------------------------------------------------------------- launcher
extern "C" void kernel_launch(void* const* d_in, const int* in_sizes, int n_in,
                              void* d_out, int out_size, void* d_ws, size_t ws_size,
                              hipStream_t stream) {
    const float* x    = (const float*)d_in[0];
    const int*   ei   = (const int*)d_in[1];
    const float* w1_0 = (const float*)d_in[2];
    const float* w1_1 = (const float*)d_in[3];
    const float* b1   = (const float*)d_in[4];
    const float* w2_0 = (const float*)d_in[5];
    const float* w2_1 = (const float*)d_in[6];
    const float* b2   = (const float*)d_in[7];
    float* out = (float*)d_out;

    int N = in_sizes[0] / FDIM;
    int E = in_sizes[1] / 2;
    const int* src = ei;
    const int* dst = ei + E;

    // workspace layout: dinv[N] | agg1[N*128] | h[N*128] | agg2[N*128] | norm[E]
    float* ws   = (float*)d_ws;
    float* dinv = ws;
    float* agg1 = dinv + N;
    float* h    = agg1 + (size_t)N * FDIM;
    float* agg2 = h + (size_t)N * FDIM;
    float* nrm  = agg2 + (size_t)N * FDIM;

    // zero atomic targets (dinv+agg1 contiguous; agg2 separate)
    cheb_zero_kernel<<<2048, 256, 0, stream>>>(dinv, (size_t)N + (size_t)N * FDIM);
    cheb_zero_kernel<<<2048, 256, 0, stream>>>(agg2, (size_t)N * FDIM);

    cheb_deg_kernel<<<(E + 255) / 256, 256, 0, stream>>>(src, dinv, E);
    cheb_dinv_kernel<<<(N + 255) / 256, 256, 0, stream>>>(dinv, N);
    cheb_norm_kernel<<<(E + 255) / 256, 256, 0, stream>>>(src, dst, dinv, nrm, E);

    {   // layer 1 aggregation: one wave per edge
        long long tw = (long long)E * 32;
        cheb_agg_kernel<<<(unsigned)((tw + 255) / 256), 256, 0, stream>>>(
            x, src, dst, nrm, agg1, E);
    }

    int nTiles = (N + 15) / 16;
    cheb_gemm_relu_wmma<<<(nTiles + 7) / 8, 256, 0, stream>>>(
        x, agg1, w1_0, w1_1, b1, h, nTiles);

    {   // layer 2 aggregation over h
        long long tw = (long long)E * 32;
        cheb_agg_kernel<<<(unsigned)((tw + 255) / 256), 256, 0, stream>>>(
            h, src, dst, nrm, agg2, E);
    }

    {   // head: dot products + sigmoid, one wave per node
        long long tw = (long long)N * 32;
        cheb_out_kernel<<<(unsigned)((tw + 255) / 256), 256, 0, stream>>>(
            h, agg2, w2_0, w2_1, b2, out, N);
    }
}
